// WindowedAttentionBlock_54846732369965
// MI455X (gfx1250) — compile-verified, matrix operand order checked by reading, not run on hardware
//
#include <hip/hip_runtime.h>
#include <math.h>

typedef __attribute__((ext_vector_type(16))) __bf16 v16bf;
typedef __attribute__((ext_vector_type(8)))  float  v8f;

#define CDIM   256
#define HDIM   220
#define HWSZ   48400      // 220*220
#define NWROW  28
#define NWIN   3136       // 4*28*28
#define NTOK   193600     // 4*220*220

// ---------- helpers ----------
static __device__ __forceinline__ unsigned short f2bf_u(float f) {
  unsigned int u = __builtin_bit_cast(unsigned int, f);
  u += 0x7fffu + ((u >> 16) & 1u);           // round-to-nearest-even
  return (unsigned short)(u >> 16);
}
static __device__ __forceinline__ __bf16 u2bf(unsigned short u) {
  return __builtin_bit_cast(__bf16, u);
}

static __device__ __forceinline__ v8f wmma_bf16(v16bf a, v16bf b, v8f c) {
  return __builtin_amdgcn_wmma_f32_16x16x32_bf16(false, a, false, b, (short)0, c, false, false);
}

// A/B fragment from row-major [rows][stride] matrix, element (row0+lane%16, col0+k)
// K packing per ISA 7.12.2: VGPR v<4 -> k = (lane>=16?8:0)+2v ; v>=4 -> k = 16+(lane>=16?8:0)+2(v-4)
// Per lane this is two contiguous 16-byte runs -> lowers to b128 loads.
static __device__ __forceinline__ v16bf load_frag(const unsigned short* p, int row0, int col0,
                                                  int stride, int lane) {
  v16bf f;
  const unsigned short* rp = p + (size_t)(row0 + (lane & 15)) * stride + col0 + ((lane >> 4) << 3);
#pragma unroll
  for (int v = 0; v < 4; ++v) { f[2*v]   = u2bf(rp[2*v]);    f[2*v+1]   = u2bf(rp[2*v+1]); }
#pragma unroll
  for (int v = 0; v < 4; ++v) { f[8+2*v] = u2bf(rp[16+2*v]); f[8+2*v+1] = u2bf(rp[16+2*v+1]); }
  return f;
}

static __device__ __forceinline__ float hmax16(float v) {
  v = fmaxf(v, __shfl_xor(v, 1, 32));
  v = fmaxf(v, __shfl_xor(v, 2, 32));
  v = fmaxf(v, __shfl_xor(v, 4, 32));
  v = fmaxf(v, __shfl_xor(v, 8, 32));
  return v;
}
static __device__ __forceinline__ float hsum16(float v) {
  v += __shfl_xor(v, 1, 32); v += __shfl_xor(v, 2, 32);
  v += __shfl_xor(v, 4, 32); v += __shfl_xor(v, 8, 32);
  return v;
}
static __device__ __forceinline__ float hsum32(float v) {
  v += __shfl_xor(v, 1, 32); v += __shfl_xor(v, 2, 32);
  v += __shfl_xor(v, 4, 32); v += __shfl_xor(v, 8, 32);
  v += __shfl_xor(v, 16, 32);
  return v;
}

// ---------- kernel 1: weights f32 -> bf16 ----------
__global__ void swin_prep_kernel(const float* __restrict__ wqkv, const float* __restrict__ wout,
                                 const float* __restrict__ w1,   const float* __restrict__ w2,
                                 unsigned short* __restrict__ ws) {
  int i = blockIdx.x * 256 + threadIdx.x;
  float v;
  if      (i < 196608) v = wqkv[i];
  else if (i < 262144) v = wout[i - 196608];
  else if (i < 524288) v = w1[i - 262144];
  else                 v = w2[i - 524288];
  ws[i] = f2bf_u(v);
}

// ---------- kernel 2: fused LN1 + windowed MHA (one workgroup per 8x8 window) ----------
__global__ void swin_attn_kernel(const float* __restrict__ x,
                                 const unsigned short* __restrict__ Wqkv,
                                 const unsigned short* __restrict__ Wout,
                                 const float* __restrict__ bqkv, const float* __restrict__ bout,
                                 const float* __restrict__ lnw,  const float* __restrict__ lnb,
                                 const float* __restrict__ gamma1,
                                 float* __restrict__ out) {
  extern __shared__ unsigned short lds[];
  unsigned short* lX = lds;                 // [64][256] LN'd window input (reused as attn-out)
  unsigned short* lQ = lds + 16384;         // [64][256]  (token-major)
  unsigned short* lK = lds + 2 * 16384;     // [64][256]  (token-major)
  unsigned short* lV = lds + 3 * 16384;     // [256][64]  V stored TRANSPOSED (channel-major)
  unsigned short* lP = lds + 4 * 16384;     // [8][64][64] probs
  unsigned short* lO = lX;

  int tid = threadIdx.x, lane = tid & 31, wave = tid >> 5;
  int wd = blockIdx.x;
  int bb = wd / 784; int rem = wd - bb * 784;
  int wh = rem / NWROW, ww = rem - wh * NWROW;

  // ---- LN1 over channels, straight into LDS (zero-fill padded tokens) ----
  for (int it = 0; it < 8; ++it) {
    int t = wave * 8 + it;
    int hh = wh * 8 + (t >> 3), wc = ww * 8 + (t & 7);
    if (hh >= HDIM || wc >= HDIM) {               // pad-after-LN => exact zeros
#pragma unroll
      for (int j = 0; j < 8; ++j) lX[t * 256 + j * 32 + lane] = 0;
      continue;
    }
    const float* px = x + (size_t)bb * CDIM * HWSZ + (size_t)hh * HDIM + wc;
    float vals[8]; float s = 0.f;
#pragma unroll
    for (int j = 0; j < 8; ++j) { int c = j * 32 + lane; vals[j] = px[(size_t)c * HWSZ]; s += vals[j]; }
    s = hsum32(s);
    float mu = s * (1.f / 256.f);
    float s2 = 0.f;
#pragma unroll
    for (int j = 0; j < 8; ++j) { float d = vals[j] - mu; s2 += d * d; }
    s2 = hsum32(s2);
    float rs = rsqrtf(s2 * (1.f / 256.f) + 1e-5f);
#pragma unroll
    for (int j = 0; j < 8; ++j) {
      int c = j * 32 + lane;
      lX[t * 256 + c] = f2bf_u((vals[j] - mu) * rs * lnw[c] + lnb[c]);
    }
  }
  __syncthreads();

  // ---- QKV projection: [64,256] x [256,768]^T, 192 tiles over 8 waves ----
  for (int tile = wave; tile < 192; tile += 8) {
    int mt = tile / 48, nt = tile - mt * 48;
    v8f acc = {};
#pragma unroll
    for (int ks = 0; ks < 8; ++ks) {
      v16bf a = load_frag(lX,   mt * 16, ks * 32, 256, lane);
      v16bf b = load_frag(Wqkv, nt * 16, ks * 32, 256, lane);   // B[k][n] = W[n][k]
      acc = wmma_bf16(a, b, acc);
    }
    int col = nt * 16 + (lane & 15);
    float bias = bqkv[col];
    if (col < 512) {
      unsigned short* dst = (col < 256) ? lQ : lK;
      int c = col & 255;
#pragma unroll
      for (int v = 0; v < 8; ++v) {
        int m = mt * 16 + v + ((lane >> 4) << 3);
        dst[m * 256 + c] = f2bf_u(acc[v] + bias);
      }
    } else {                                  // V: store transposed [channel][token]
      int c = col - 512;
#pragma unroll
      for (int v = 0; v < 8; ++v) {
        int m = mt * 16 + v + ((lane >> 4) << 3);
        lV[c * 64 + m] = f2bf_u(acc[v] + bias);
      }
    }
  }
  __syncthreads();

  // ---- scores + softmax: wave = head ----
  const float scale = 0.17677669529663687f;   // 32^-0.5
  for (int mt = 0; mt < 4; ++mt) {
    v16bf aq = load_frag(lQ, mt * 16, wave * 32, 256, lane);
    v8f sf[4];
#pragma unroll
    for (int nt = 0; nt < 4; ++nt) {
      v8f acc = {};
      v16bf bk = load_frag(lK, nt * 16, wave * 32, 256, lane);  // B[d][n] = K[n][d]
      sf[nt] = wmma_bf16(aq, bk, acc);
    }
#pragma unroll
    for (int v = 0; v < 8; ++v) {
      float mx = -3.4e38f;
#pragma unroll
      for (int nt = 0; nt < 4; ++nt) { sf[nt][v] *= scale; mx = fmaxf(mx, sf[nt][v]); }
      mx = hmax16(mx);
      float sum = 0.f;
#pragma unroll
      for (int nt = 0; nt < 4; ++nt) { float e = __expf(sf[nt][v] - mx); sf[nt][v] = e; sum += e; }
      sum = hsum16(sum);
      float inv = 1.f / sum;
      int m = mt * 16 + v + ((lane >> 4) << 3);
#pragma unroll
      for (int nt = 0; nt < 4; ++nt)
        lP[(size_t)(wave * 64 + m) * 64 + nt * 16 + (lane & 15)] = f2bf_u(sf[nt][v] * inv);
    }
  }
  // each wave consumes only its own probs; V was barriered above
  // ---- attn @ V: [64,64] x [64,32] per head (V^T in LDS -> contiguous B frags) ----
  for (int mt = 0; mt < 4; ++mt) {
#pragma unroll
    for (int nt2 = 0; nt2 < 2; ++nt2) {
      v8f acc = {};
#pragma unroll
      for (int ks = 0; ks < 2; ++ks) {
        v16bf a = load_frag(lP + wave * 4096, mt * 16, ks * 32, 64, lane);
        v16bf b = load_frag(lV, wave * 32 + nt2 * 16, ks * 32, 64, lane); // B[k][n] = Vt[n][k]
        acc = wmma_bf16(a, b, acc);
      }
      int c = wave * 32 + nt2 * 16 + (lane & 15);
#pragma unroll
      for (int v = 0; v < 8; ++v) {
        int m = mt * 16 + v + ((lane >> 4) << 3);
        lO[m * 256 + c] = f2bf_u(acc[v]);
      }
    }
  }
  __syncthreads();

  // ---- out projection + residual: x1 = x + g1*(attn@Wout^T + b) ----
  float g1 = gamma1[0];
  for (int tile = wave; tile < 64; tile += 8) {
    int mt = tile >> 4, nt = tile & 15;
    v8f acc = {};
#pragma unroll
    for (int ks = 0; ks < 8; ++ks) {
      v16bf a = load_frag(lO,   mt * 16, ks * 32, 256, lane);
      v16bf b = load_frag(Wout, nt * 16, ks * 32, 256, lane);
      acc = wmma_bf16(a, b, acc);
    }
    int c = nt * 16 + (lane & 15);
    float bias = bout[c];
#pragma unroll
    for (int v = 0; v < 8; ++v) {
      int m  = mt * 16 + v + ((lane >> 4) << 3);
      int hh = wh * 8 + (m >> 3), wc = ww * 8 + (m & 7);
      if (hh < HDIM && wc < HDIM) {
        size_t gi = ((size_t)(bb * CDIM + c) * HDIM + hh) * HDIM + wc;
        out[gi] = x[gi] + g1 * (acc[v] + bias);
      }
    }
  }
}

// ---------- kernel 3: LN2 + FFN + residual, in place on d_out ----------
__global__ void swin_ffn_kernel(const unsigned short* __restrict__ W1,
                                const unsigned short* __restrict__ W2,
                                const float* __restrict__ b1, const float* __restrict__ b2,
                                const float* __restrict__ lnw, const float* __restrict__ lnb,
                                const float* __restrict__ gamma2, float* __restrict__ out) {
  extern __shared__ unsigned short lds[];
  unsigned short* lX = lds;            // [64][256]  LN2(x1) bf16
  unsigned short* lH = lds + 16384;    // [64][1024] gelu hidden bf16

  int tid = threadIdx.x, lane = tid & 31, wave = tid >> 5;
  int base = blockIdx.x * 64;

  // LN2 over channels
  for (int it = 0; it < 8; ++it) {
    int t = wave * 8 + it, tt = base + t;
    int b = tt / HWSZ; int r = tt - b * HWSZ;
    int hh = r / HDIM, wc = r - hh * HDIM;
    const float* px = out + (size_t)b * CDIM * HWSZ + (size_t)hh * HDIM + wc;
    float vals[8]; float s = 0.f;
#pragma unroll
    for (int j = 0; j < 8; ++j) { int c = j * 32 + lane; vals[j] = px[(size_t)c * HWSZ]; s += vals[j]; }
    s = hsum32(s);
    float mu = s * (1.f / 256.f);
    float s2 = 0.f;
#pragma unroll
    for (int j = 0; j < 8; ++j) { float d = vals[j] - mu; s2 += d * d; }
    s2 = hsum32(s2);
    float rs = rsqrtf(s2 * (1.f / 256.f) + 1e-5f);
#pragma unroll
    for (int j = 0; j < 8; ++j) {
      int c = j * 32 + lane;
      lX[t * 256 + c] = f2bf_u((vals[j] - mu) * rs * lnw[c] + lnb[c]);
    }
  }
  __syncthreads();

  // GEMM1 [64,256]x[256,1024]^T + GELU(exact)
  for (int tile = wave; tile < 256; tile += 8) {
    int mt = tile >> 6, nt = tile & 63;
    v8f acc = {};
#pragma unroll
    for (int ks = 0; ks < 8; ++ks) {
      v16bf a = load_frag(lX, mt * 16, ks * 32, 256, lane);
      v16bf b = load_frag(W1, nt * 16, ks * 32, 256, lane);
      acc = wmma_bf16(a, b, acc);
    }
    int col = nt * 16 + (lane & 15);
    float bias = b1[col];
#pragma unroll
    for (int v = 0; v < 8; ++v) {
      int m = mt * 16 + v + ((lane >> 4) << 3);
      float h = acc[v] + bias;
      h = 0.5f * h * (1.f + erff(h * 0.70710678118654752f));
      lH[m * 1024 + col] = f2bf_u(h);
    }
  }
  __syncthreads();

  // GEMM2 [64,1024]x[1024,256]^T + residual
  float g2 = gamma2[0];
  for (int tile = wave; tile < 64; tile += 8) {
    int mt = tile >> 4, nt = tile & 15;
    v8f acc = {};
#pragma unroll
    for (int ks = 0; ks < 32; ++ks) {
      v16bf a = load_frag(lH, mt * 16, ks * 32, 1024, lane);
      v16bf b = load_frag(W2, nt * 16, ks * 32, 1024, lane);
      acc = wmma_bf16(a, b, acc);
    }
    int c = nt * 16 + (lane & 15);
    float bias = b2[c];
#pragma unroll
    for (int v = 0; v < 8; ++v) {
      int m = mt * 16 + v + ((lane >> 4) << 3);
      int tt = base + m;
      int b = tt / HWSZ; int r = tt - b * HWSZ;
      int hh = r / HDIM, wc = r - hh * HDIM;
      size_t gi = ((size_t)(b * CDIM + c) * HDIM + hh) * HDIM + wc;
      out[gi] = out[gi] + g2 * (acc[v] + bias);
    }
  }
}

// ---------- launcher ----------
extern "C" void kernel_launch(void* const* d_in, const int* in_sizes, int n_in,
                              void* d_out, int out_size, void* d_ws, size_t ws_size,
                              hipStream_t stream) {
  const float* x          = (const float*)d_in[0];
  const float* in_proj_w  = (const float*)d_in[1];
  const float* in_proj_b  = (const float*)d_in[2];
  const float* out_proj_w = (const float*)d_in[3];
  const float* out_proj_b = (const float*)d_in[4];
  const float* ln1_w      = (const float*)d_in[5];
  const float* ln1_b      = (const float*)d_in[6];
  const float* ln2_w      = (const float*)d_in[7];
  const float* ln2_b      = (const float*)d_in[8];
  const float* ffn_w1     = (const float*)d_in[9];
  const float* ffn_b1     = (const float*)d_in[10];
  const float* ffn_w2     = (const float*)d_in[11];
  const float* ffn_b2     = (const float*)d_in[12];
  const float* gamma1     = (const float*)d_in[13];
  const float* gamma2     = (const float*)d_in[14];
  float* out = (float*)d_out;

  unsigned short* ws   = (unsigned short*)d_ws;
  unsigned short* Wqkv = ws;                 // 768*256
  unsigned short* Wout = ws + 196608;        // 256*256
  unsigned short* W1b  = ws + 262144;        // 1024*256
  unsigned short* W2b  = ws + 524288;        // 256*1024

  swin_prep_kernel<<<3072, 256, 0, stream>>>(in_proj_w, out_proj_w, ffn_w1, ffn_w2, ws);
  swin_attn_kernel<<<NWIN, 256, 196608, stream>>>(x, Wqkv, Wout, in_proj_b, out_proj_b,
                                                  ln1_w, ln1_b, gamma1, out);
  swin_ffn_kernel<<<NTOK / 64, 256, 163840, stream>>>(W1b, W2b, ffn_b1, ffn_b2,
                                                      ln2_w, ln2_b, gamma2, out);
}
